// MetaGraphNetWork_84284438217072
// MI455X (gfx1250) — matrix operand order repaired and checked by reading, not compiled, not run on hardware
//
#include <hip/hip_runtime.h>

// MetaGraphNetWork forward for MI455X (gfx1250, wave32, WMMA f16->f32,
// async direct-to-LDS gather for edge feature tiles).
#define N_E   40000
#define N_R   400
#define HD    128
#define EE    300000
#define KIN   512
#define LN_EPS 1e-5f

typedef __attribute__((ext_vector_type(16))) _Float16 v16h;
typedef __attribute__((ext_vector_type(8)))  _Float16 v8h;
typedef __attribute__((ext_vector_type(8)))  float    v8f;

static __device__ __forceinline__ v8f wmma32(v16h a, v16h b, v8f c) {
  return __builtin_amdgcn_wmma_f32_16x16x32_f16(false, a, false, b, (short)0, c, false, false);
}

// A-frag (16x32 f16) from row-major LDS tile. Lane layout per ISA 7.12.2:
// lanes 0-15: K = k0+0..7 and k0+16..23 ; lanes 16-31: K = k0+8..15 and k0+24..31 ; M = lane&15.
static __device__ __forceinline__ v16h frag_a_lds(const _Float16* tile, int stride, int k0) {
  const int l = threadIdx.x & 31;
  const _Float16* p = tile + (l & 15) * stride + k0 + ((l >> 4) << 3);
  v8h lo = *(const v8h*)p;
  v8h hi = *(const v8h*)(p + 16);
  return __builtin_shufflevector(lo, hi, 0,1,2,3,4,5,6,7,8,9,10,11,12,13,14,15);
}

// B-frag (32x16 f16) from LDS weight stored TRANSPOSED as WT[n][k] (stride K).
// lanes 0-15: K = k0+0..7 and k0+16..23 ; lanes 16-31: K = k0+8..15 and k0+24..31 ; N = lane&15.
static __device__ __forceinline__ v16h frag_b_ldsT(const _Float16* wT, int K, int n0, int k0) {
  const int l = threadIdx.x & 31;
  const _Float16* p = wT + (n0 + (l & 15)) * K + k0 + ((l >> 4) << 3);
  v8h lo = *(const v8h*)p;
  v8h hi = *(const v8h*)(p + 16);
  return __builtin_shufflevector(lo, hi, 0,1,2,3,4,5,6,7,8,9,10,11,12,13,14,15);
}

// A-frag directly from global f32 row (per-lane row pointer), scaled + cvt to f16.
static __device__ __forceinline__ v16h frag_a_gf32(const float* rowbase, int k0, float scale) {
  const int l = threadIdx.x & 31;
  const float* p = rowbase + k0 + ((l >> 4) << 3);
  v16h a;
#pragma unroll
  for (int i = 0; i < 8; ++i) a[i] = (_Float16)(p[i] * scale);
#pragma unroll
  for (int i = 0; i < 8; ++i) a[8 + i] = (_Float16)(p[16 + i] * scale);
  return a;
}

// ---------------- degree: deg[dst] += 1 (same for both layers) ----------------
__global__ void deg_kernel(const int* __restrict__ dst, int n, float* __restrict__ deg) {
  int i = blockIdx.x * blockDim.x + threadIdx.x;
  if (i < n) unsafeAtomicAdd(&deg[dst[i]], 1.0f);
}

// ------------- fused projection: out = LN(relu(X @ W + b)) -------------------
// X:[M,K] f32, W:[K,128] f32 -> f16 LDS transposed. Writes f32 + f16 copies.
__global__ void proj_ln_kernel(const float* __restrict__ X, const float* __restrict__ W,
                               const float* __restrict__ bias, const float* __restrict__ g,
                               const float* __restrict__ beta, int M, int K,
                               float* __restrict__ outF, _Float16* __restrict__ outH) {
  extern __shared__ char smem[];
  _Float16* WT = (_Float16*)smem;  // [128][K]
  const int tid = threadIdx.x;
  for (int idx = tid; idx < K * HD; idx += blockDim.x) {
    int k = idx >> 7, n = idx & 127;
    WT[n * K + k] = (_Float16)W[idx];
  }
  __syncthreads();
  const int l = tid & 31, wave = tid >> 5, nwaves = blockDim.x >> 5;
  const int n = l & 15;
  float biasv[8], gv[8], betav[8];
#pragma unroll
  for (int nt = 0; nt < 8; ++nt) {
    biasv[nt] = bias[nt * 16 + n];
    gv[nt] = g[nt * 16 + n];
    betav[nt] = beta[nt * 16 + n];
  }
  const int ntiles = M >> 4;
  for (int tile = blockIdx.x * nwaves + wave; tile < ntiles; tile += gridDim.x * nwaves) {
    const float* rowp = X + (size_t)((tile << 4) + n) * K;
    v8f c[8];
#pragma unroll
    for (int nt = 0; nt < 8; ++nt)
#pragma unroll
      for (int r = 0; r < 8; ++r) c[nt][r] = biasv[nt];
    for (int ks = 0; ks < K; ks += 32) {
      v16h a = frag_a_gf32(rowp, ks, 1.0f);
#pragma unroll
      for (int nt = 0; nt < 8; ++nt) c[nt] = wmma32(a, frag_b_ldsT(WT, K, nt * 16, ks), c[nt]);
    }
#pragma unroll
    for (int nt = 0; nt < 8; ++nt)
#pragma unroll
      for (int r = 0; r < 8; ++r) c[nt][r] = fmaxf(c[nt][r], 0.0f);
    // LayerNorm per row (row m = r + 8*(lane>=16); reduce across the 16 lanes of the group)
#pragma unroll
    for (int r = 0; r < 8; ++r) {
      float s = 0.f, sq = 0.f;
#pragma unroll
      for (int nt = 0; nt < 8; ++nt) { float v = c[nt][r]; s += v; sq += v * v; }
      s += __shfl_xor(s, 1, 32);  sq += __shfl_xor(sq, 1, 32);
      s += __shfl_xor(s, 2, 32);  sq += __shfl_xor(sq, 2, 32);
      s += __shfl_xor(s, 4, 32);  sq += __shfl_xor(sq, 4, 32);
      s += __shfl_xor(s, 8, 32);  sq += __shfl_xor(sq, 8, 32);
      float mean = s * (1.0f / HD);
      float var = sq * (1.0f / HD) - mean * mean;
      float rstd = rsqrtf(var + LN_EPS);
      int mrow = (tile << 4) + r + ((l >> 4) << 3);
#pragma unroll
      for (int nt = 0; nt < 8; ++nt) {
        int col = nt * 16 + n;
        float v = (c[nt][r] - mean) * rstd * gv[nt] + betav[nt];
        outF[(size_t)mrow * HD + col] = v;
        outH[(size_t)mrow * HD + col] = (_Float16)v;
      }
    }
  }
}

// --------- edge kernel: MLP weight + sigmoid + weighted scatter-add -----------
// Per 16-edge tile (one wave): A = [hs|hd] 16x256 f16 gathered straight into LDS
// via the gfx1250 async direct-to-LDS path (ASYNCcnt), then
// z = relu(A@W1+b1), dw = sigmoid(z.W2+b2), ada += dw^2, S[dst] += hs*(w+dw).
#define ASYNC_B128(OFF)                                                     \
  asm volatile("global_load_async_to_lds_b128 %0, %1, off offset:" #OFF    \
               :: "v"(ldsoff), "v"(ga) : "memory")

__global__ void edge_kernel(const _Float16* __restrict__ hsrc, const _Float16* __restrict__ hdst,
                            const int* __restrict__ src, const int* __restrict__ dst,
                            const float* __restrict__ wbase,
                            const float* __restrict__ W1, const float* __restrict__ b1,
                            const float* __restrict__ W2, const float* __restrict__ b2p,
                            float* __restrict__ S, float* __restrict__ ada) {
  extern __shared__ char smem[];
  _Float16* W1T = (_Float16*)smem;                     // [128][256] transposed, 64KB
  float* b1s = (float*)(smem + 128 * 256 * 2);
  float* W2s = b1s + 128;
  _Float16* Abase = (_Float16*)(smem + 128 * 256 * 2 + 1024);  // per-wave 16x256 f16
  const int tid = threadIdx.x;
  for (int idx = tid; idx < 256 * 128; idx += blockDim.x) {
    int k = idx >> 7, n = idx & 127;
    W1T[n * 256 + k] = (_Float16)W1[idx];
  }
  for (int idx = tid; idx < 128; idx += blockDim.x) { b1s[idx] = b1[idx]; W2s[idx] = W2[idx]; }
  __syncthreads();
  const int l = tid & 31, wave = tid >> 5, nwaves = blockDim.x >> 5;
  const int m = l & 15, p = l >> 4;
  _Float16* At = Abase + wave * (16 * 256);
  const float b2v = b2p[0];
  float b1v[8], w2v[8];
#pragma unroll
  for (int nt = 0; nt < 8; ++nt) { b1v[nt] = b1s[nt * 16 + m]; w2v[nt] = W2s[nt * 16 + m]; }
  float adacc = 0.0f;
  const int ntiles = EE >> 4;
  for (int tile = blockIdx.x * nwaves + wave; tile < ntiles; tile += gridDim.x * nwaves) {
    const int e0 = tile << 4;
    {  // async gather: lane (m,p) pulls 128 f16 of row m (p=0: hs, p=1: hd) into LDS
      int eid = e0 + m;
      int node = (p == 0) ? src[eid] : dst[eid];
      const _Float16* grow = ((p == 0) ? hsrc : hdst) + (size_t)node * HD;
      unsigned long long ga = (unsigned long long)(size_t)grow;
      unsigned ldsoff = (unsigned)(size_t)(At + m * 256 + p * 128);  // flat->LDS: addr[31:0]
      // prior-iteration LDS readers (frag/scatter ds loads) must drain before overwrite
      asm volatile("s_wait_dscnt 0x0" ::: "memory");
      ASYNC_B128(0);   ASYNC_B128(16);  ASYNC_B128(32);  ASYNC_B128(48);
      ASYNC_B128(64);  ASYNC_B128(80);  ASYNC_B128(96);  ASYNC_B128(112);
      ASYNC_B128(128); ASYNC_B128(144); ASYNC_B128(160); ASYNC_B128(176);
      ASYNC_B128(192); ASYNC_B128(208); ASYNC_B128(224); ASYNC_B128(240);
      asm volatile("s_wait_asynccnt 0x0" ::: "memory");
    }
    __builtin_amdgcn_wave_barrier();
    v16h a8[8];
#pragma unroll
    for (int ks = 0; ks < 8; ++ks) a8[ks] = frag_a_lds(At, 256, ks * 32);
    float pl[8];
#pragma unroll
    for (int r = 0; r < 8; ++r) pl[r] = 0.f;
#pragma unroll
    for (int nt = 0; nt < 8; ++nt) {
      v8f c;
#pragma unroll
      for (int r = 0; r < 8; ++r) c[r] = b1v[nt];
#pragma unroll
      for (int ks = 0; ks < 8; ++ks)
        c = wmma32(a8[ks], frag_b_ldsT(W1T, 256, nt * 16, ks * 32), c);
#pragma unroll
      for (int r = 0; r < 8; ++r) pl[r] += fmaxf(c[r], 0.f) * w2v[nt];
    }
#pragma unroll
    for (int r = 0; r < 8; ++r) {
      pl[r] += __shfl_xor(pl[r], 1, 32);
      pl[r] += __shfl_xor(pl[r], 2, 32);
      pl[r] += __shfl_xor(pl[r], 4, 32);
      pl[r] += __shfl_xor(pl[r], 8, 32);
    }
#pragma unroll
    for (int r = 0; r < 8; ++r) {
      float dw = 1.0f / (1.0f + __expf(-(pl[r] + b2v)));
      if (m == 0) adacc += dw * dw;  // one lane per 16-lane group; 8 rows each
      int m2 = r + (p << 3);
      int eid = e0 + m2;
      float ew = wbase[eid] + dw;
      float* Srow = S + (size_t)dst[eid] * HD;
      const _Float16* hsrow = At + m2 * 256;  // hs = cols 0..127
#pragma unroll
      for (int j = 0; j < 8; ++j) {
        int k = m + (j << 4);  // coalesced across the 16 lanes of the group
        unsafeAtomicAdd(&Srow[k], (float)hsrow[k] * ew);
      }
    }
  }
  if ((tid & 15) == 0) unsafeAtomicAdd(ada, adacc);
}

// ---- fused: out = relu(S0/clip(d0)@Wc0 + S1/clip(d1)@Wc1 + H@Wl + bias) -----
__global__ void norm_gemm_kernel(const float* __restrict__ S0, const float* __restrict__ deg0,
                                 const float* __restrict__ S1, const float* __restrict__ deg1,
                                 const float* __restrict__ Hc,
                                 const float* __restrict__ Wc0, const float* __restrict__ Wc1,
                                 const float* __restrict__ Wl, const float* __restrict__ biasp,
                                 int M, float* __restrict__ outF, _Float16* __restrict__ outH) {
  extern __shared__ char smem[];
  _Float16* T0 = (_Float16*)smem;
  _Float16* T1 = T0 + 128 * 128;
  _Float16* T2 = T1 + 128 * 128;
  const int tid = threadIdx.x;
  for (int idx = tid; idx < 128 * 128; idx += blockDim.x) {
    int k = idx >> 7, n = idx & 127;
    T0[n * 128 + k] = (_Float16)Wc0[idx];
    T1[n * 128 + k] = (_Float16)Wc1[idx];
    T2[n * 128 + k] = (_Float16)Wl[idx];
  }
  __syncthreads();
  const int l = tid & 31, wave = tid >> 5, nwaves = blockDim.x >> 5;
  const int n = l & 15;
  float biasv[8];
#pragma unroll
  for (int nt = 0; nt < 8; ++nt) biasv[nt] = biasp[nt * 16 + n];
  const int ntiles = M >> 4;
  for (int tile = blockIdx.x * nwaves + wave; tile < ntiles; tile += gridDim.x * nwaves) {
    const int row = (tile << 4) + n;
    const float inv0 = 1.0f / fmaxf(deg0[row], 1.0f);
    const float inv1 = 1.0f / fmaxf(deg1[row], 1.0f);
    const float* r0 = S0 + (size_t)row * HD;
    const float* r1 = S1 + (size_t)row * HD;
    const float* r2 = Hc + (size_t)row * HD;
    v8f c[8];
#pragma unroll
    for (int nt = 0; nt < 8; ++nt)
#pragma unroll
      for (int r = 0; r < 8; ++r) c[nt][r] = biasv[nt];
#pragma unroll
    for (int ks = 0; ks < 4; ++ks) {
      v16h a0 = frag_a_gf32(r0, ks * 32, inv0);
      v16h a1 = frag_a_gf32(r1, ks * 32, inv1);
      v16h a2 = frag_a_gf32(r2, ks * 32, 1.0f);
#pragma unroll
      for (int nt = 0; nt < 8; ++nt) {
        c[nt] = wmma32(a0, frag_b_ldsT(T0, 128, nt * 16, ks * 32), c[nt]);
        c[nt] = wmma32(a1, frag_b_ldsT(T1, 128, nt * 16, ks * 32), c[nt]);
        c[nt] = wmma32(a2, frag_b_ldsT(T2, 128, nt * 16, ks * 32), c[nt]);
      }
    }
#pragma unroll
    for (int r = 0; r < 8; ++r) {
      int mrow = (tile << 4) + r + ((l >> 4) << 3);
#pragma unroll
      for (int nt = 0; nt < 8; ++nt) {
        float v = fmaxf(c[nt][r], 0.f);
        int col = nt * 16 + n;
        outF[(size_t)mrow * HD + col] = v;
        if (outH) outH[(size_t)mrow * HD + col] = (_Float16)v;
      }
    }
  }
}

__global__ void finalize_kernel(const float* __restrict__ ada, float* __restrict__ out) {
  if (threadIdx.x == 0 && blockIdx.x == 0) {
    float s = 0.f;
    for (int i = 0; i < 8; ++i) s += sqrtf(ada[i] / (float)EE);
    out[0] = s / 4.0f / 2.0f * 0.05f;  // /4 /L * ADA_LOSS_W
  }
}

extern "C" void kernel_launch(void* const* d_in, const int* in_sizes, int n_in,
                              void* d_out, int out_size, void* d_ws, size_t ws_size,
                              hipStream_t stream) {
  (void)in_sizes; (void)n_in; (void)out_size; (void)ws_size;
  const float* rel_embeds = (const float*)d_in[0];
  const float* ent_embeds = (const float*)d_in[1];
  const int* srcs[4] = {(const int*)d_in[2], (const int*)d_in[5], (const int*)d_in[8], (const int*)d_in[11]};
  const int* dsts[4] = {(const int*)d_in[3], (const int*)d_in[6], (const int*)d_in[9], (const int*)d_in[12]};
  const float* wbs[4] = {(const float*)d_in[4], (const float*)d_in[7], (const float*)d_in[10], (const float*)d_in[13]};
  const float* rel_proj_W = (const float*)d_in[14];
  const float* rel_proj_b = (const float*)d_in[15];
  const float* rel_ln_g   = (const float*)d_in[16];
  const float* rel_ln_b   = (const float*)d_in[17];
  const float* ent_proj_W = (const float*)d_in[18];
  const float* ent_proj_b = (const float*)d_in[19];
  const float* ent_ln_g   = (const float*)d_in[20];
  const float* ent_ln_b   = (const float*)d_in[21];
  const float* conv_w     = (const float*)d_in[22];  // [2,4,128,128]
  const float* loop_w     = (const float*)d_in[23];  // [2,128,128]
  const float* h_bias     = (const float*)d_in[24];  // [2,128]
  const float* wp_W1      = (const float*)d_in[25];  // [2,4,256,128]
  const float* wp_b1      = (const float*)d_in[26];  // [2,4,128]
  const float* wp_W2      = (const float*)d_in[27];  // [2,4,128]
  const float* wp_b2      = (const float*)d_in[28];  // [2,4]

  char* ws = (char*)d_ws;
  size_t off = 0;
  auto take = [&](size_t bytes) -> char* {
    char* p = ws + off;
    off = (off + bytes + 255) & ~(size_t)255;
    return p;
  };
  float* heA = (float*)take((size_t)N_E * HD * 4);
  float* heB = (float*)take((size_t)N_E * HD * 4);
  float* hrA = (float*)take((size_t)N_R * HD * 4);
  float* hrB = (float*)take((size_t)N_R * HD * 4);
  _Float16* he16 = (_Float16*)take((size_t)N_E * HD * 2);
  _Float16* hr16 = (_Float16*)take((size_t)N_R * HD * 2);
  float* S[4];
  size_t Sbytes[4] = {(size_t)N_E * HD * 4, (size_t)N_E * HD * 4, (size_t)N_R * HD * 4, (size_t)N_R * HD * 4};
  for (int i = 0; i < 4; ++i) S[i] = (float*)take(Sbytes[i]);
  float* deg[4];
  size_t dbytes[4] = {(size_t)N_E * 4, (size_t)N_E * 4, (size_t)N_R * 4, (size_t)N_R * 4};
  for (int i = 0; i < 4; ++i) deg[i] = (float*)take(dbytes[i]);
  float* ada = (float*)take(8 * 4);

  const int EDGE_SMEM = 128 * 256 * 2 + 1024 + 8 * 16 * 256 * 2;  // 132096 B
  const int PROJ_SMEM = 128 * KIN * 2;                            // 131072 B
  const int NORM_SMEM = 3 * 128 * 128 * 2;                        // 98304 B
  const int EDGE_BLOCKS = (EE / 16 + 7) / 8;                      // 1 tile/wave

  // degrees + ada accumulators (shared across layers)
  for (int i = 0; i < 4; ++i) hipMemsetAsync(deg[i], 0, dbytes[i], stream);
  hipMemsetAsync(ada, 0, 8 * 4, stream);
  for (int i = 0; i < 4; ++i)
    deg_kernel<<<(EE + 255) / 256, 256, 0, stream>>>(dsts[i], EE, deg[i]);

  // input projections + LN (f32 + f16 copies)
  proj_ln_kernel<<<320, 256, PROJ_SMEM, stream>>>(ent_embeds, ent_proj_W, ent_proj_b,
                                                  ent_ln_g, ent_ln_b, N_E, KIN, heA, he16);
  proj_ln_kernel<<<4, 256, PROJ_SMEM, stream>>>(rel_embeds, rel_proj_W, rel_proj_b,
                                                rel_ln_g, rel_ln_b, N_R, KIN, hrA, hr16);

  const float* heCur = heA;
  const float* hrCur = hrA;
  for (int lid = 0; lid < 2; ++lid) {
    for (int i = 0; i < 4; ++i) hipMemsetAsync(S[i], 0, Sbytes[i], stream);
    const _Float16* hs16[4] = {hr16, hr16, he16, he16};
    const _Float16* hd16[4] = {he16, he16, hr16, hr16};
    for (int ri = 0; ri < 4; ++ri) {
      int e = lid * 4 + ri;
      edge_kernel<<<EDGE_BLOCKS, 256, EDGE_SMEM, stream>>>(
          hs16[ri], hd16[ri], srcs[ri], dsts[ri], wbs[ri],
          wp_W1 + (size_t)e * 256 * 128, wp_b1 + (size_t)e * 128,
          wp_W2 + (size_t)e * 128, wp_b2 + e, S[ri], ada + e);
    }
    float* outE = (lid == 0) ? heB : (float*)d_out;
    float* outR = (lid == 0) ? hrB : (float*)d_out + (size_t)N_E * HD;
    _Float16* outE16 = (lid == 0) ? he16 : nullptr;
    _Float16* outR16 = (lid == 0) ? hr16 : nullptr;
    norm_gemm_kernel<<<320, 256, NORM_SMEM, stream>>>(
        S[0], deg[0], S[1], deg[1], heCur,
        conv_w + (size_t)(lid * 4 + 0) * 128 * 128, conv_w + (size_t)(lid * 4 + 1) * 128 * 128,
        loop_w + (size_t)lid * 128 * 128, h_bias + (size_t)lid * 128, N_E, outE, outE16);
    norm_gemm_kernel<<<4, 256, NORM_SMEM, stream>>>(
        S[2], deg[2], S[3], deg[3], hrCur,
        conv_w + (size_t)(lid * 4 + 2) * 128 * 128, conv_w + (size_t)(lid * 4 + 3) * 128 * 128,
        loop_w + (size_t)lid * 128 * 128, h_bias + (size_t)lid * 128, N_R, outR, outR16);
    heCur = heB;
    hrCur = hrB;
  }
  finalize_kernel<<<1, 32, 0, stream>>>(ada, (float*)d_out + (size_t)N_E * HD + (size_t)N_R * HD);
}
